// BiaffineNERModel_42236708389785
// MI455X (gfx1250) — compile-verified, matrix operand order checked by reading, not run on hardware
//
#include <hip/hip_runtime.h>

typedef __attribute__((ext_vector_type(16))) _Float16 v16h;
typedef __attribute__((ext_vector_type(8)))  _Float16 v8h;
typedef __attribute__((ext_vector_type(8)))  float    v8f;

#define NB   16     // batch
#define NT   256    // seq len
#define ND   768    // input dim
#define NH   400    // hidden per dir
#define NH2  800    // 2H
#define NH3  1200   // 3H (gates i,j,o)
#define NHP  416    // H padded to mult of 32 (K dim of recurrent GEMM)
#define NF   150
#define NFP  160    // F+1 padded (151 -> 160)
#define NC   8
#define BTOT (NB*NT) // 4096

// ---------------------------------------------------------------------------
// Fragment helpers (16x16x32 f16 WMMA, wave32)
// A (16x32, MxK) lane layout: lanes 0-15 row M=lane, K {k0..k0+7, k0+16..k0+23};
// lanes 16-31 row M=lane-16, K {k0+8..k0+15, k0+24..k0+31}.
// B loaded from Bt = B^T stored [N, Kpad] row-major => same loader, row = n.
// ---------------------------------------------------------------------------
__device__ __forceinline__ v16h ldh16(const _Float16* p) {
  v8h lo = *(const v8h*)(p);
  v8h hi = *(const v8h*)(p + 16);
  v16h f;
#pragma unroll
  for (int i = 0; i < 8; ++i) { f[i] = lo[i]; f[i + 8] = hi[i]; }
  return f;
}

// Register-blocked MB x NBLK tile-group. LDA/LDB are compile-time so every
// fragment load is base-pointer + immediate offset (no live pointer arrays).
template <int MB, int NBLK, int LDA, int LDB>
__device__ __forceinline__ void wmma_block(const _Float16* A, int row0,
                                           const _Float16* Bt, int col0,
                                           int K, v8f (&acc)[MB][NBLK]) {
  const int lane = threadIdx.x & 31;
  const int koff = (lane >> 4) << 3;
  const _Float16* ap = A + (size_t)(row0 + (lane & 15)) * LDA + koff;
  const _Float16* bp = Bt + (size_t)(col0 + (lane & 15)) * LDB + koff;
  for (int k0 = 0; k0 < K; k0 += 32) {
    v16h a[MB];
#pragma unroll
    for (int i = 0; i < MB; ++i) a[i] = ldh16(ap + i * 16 * LDA + k0);
#pragma unroll
    for (int j = 0; j < NBLK; ++j) {
      v16h b = ldh16(bp + j * 16 * LDB + k0);
#pragma unroll
      for (int i = 0; i < MB; ++i)
        acc[i][j] = __builtin_amdgcn_wmma_f32_16x16x32_f16(false, a[i], false, b,
                                                           (short)0, acc[i][j], false, false);
    }
  }
}

// Fast transcendentals: v_rcp_f32 instead of IEEE div expansion.
__device__ __forceinline__ float rcp_(float x)  { return __builtin_amdgcn_rcpf(x); }
__device__ __forceinline__ float sigm_(float x) { return rcp_(1.f + __expf(-x)); }
__device__ __forceinline__ float tanh_(float x) { return 1.f - 2.f * rcp_(__expf(2.f * x) + 1.f); }

// ---------------------------------------------------------------------------
// Conversion / transpose kernels
// ---------------------------------------------------------------------------
__global__ void k_f32_to_f16(const float* __restrict__ src, _Float16* __restrict__ dst, int n) {
  int i = blockIdx.x * blockDim.x + threadIdx.x;
  if (i < n) dst[i] = (_Float16)src[i];
}

// Wt[n*Kpad + k] = W[(row_off+k)*srcld + n]  (zero-padded)
__global__ void k_transpose_w(const float* __restrict__ W, _Float16* __restrict__ Wt,
                              int row_off, int K, int Kpad, int N, int Npad, int srcld) {
  int idx = blockIdx.x * blockDim.x + threadIdx.x;
  if (idx >= Npad * Kpad) return;
  int n = idx / Kpad, k = idx % Kpad;
  float v = (n < N && k < K) ? W[(size_t)(row_off + k) * srcld + n] : 0.f;
  Wt[idx] = (_Float16)v;
}

// Ut[(c*160 + j)*160 + i] = U[(i*8 + c)*151 + j]  (zero-padded to 160x160)
__global__ void k_transpose_U(const float* __restrict__ U, _Float16* __restrict__ Ut) {
  int idx = blockIdx.x * blockDim.x + threadIdx.x;
  if (idx >= NC * NFP * NFP) return;
  int i = idx % NFP;
  int j = (idx / NFP) % NFP;
  int c = idx / (NFP * NFP);
  float v = (i < NF + 1 && j < NF + 1) ? U[((size_t)i * NC + c) * (NF + 1) + j] : 0.f;
  Ut[idx] = (_Float16)v;
}

// ---------------------------------------------------------------------------
// Xg precompute: C_f32[4096,1200] = A_f16[4096,K] * Bt_f16[1200,K]^T + bias
// 4x3 register blocking, compile-time geometry.
// ---------------------------------------------------------------------------
template <int KDIM, int LD>
__global__ __launch_bounds__(256, 1)
void k_gemm_bias_f32(const _Float16* __restrict__ A, const _Float16* __restrict__ Bt,
                     const float* __restrict__ bias, float* __restrict__ C) {
  constexpr int MB = 4, NBLK = 3;
  constexpr int tn = NH3 / (16 * NBLK);             // 25
  int wave = blockIdx.x * (blockDim.x >> 5) + (threadIdx.x >> 5);
  int row0 = (wave / tn) * (16 * MB);
  int col0 = (wave % tn) * (16 * NBLK);
  v8f acc[MB][NBLK] = {};
  wmma_block<MB, NBLK, LD, LD>(A, row0, Bt, col0, KDIM, acc);
  int lane = threadIdx.x & 31;
  int nl = lane & 15;
  int m0 = (lane >> 4) << 3;
#pragma unroll
  for (int i = 0; i < MB; ++i)
#pragma unroll
    for (int j = 0; j < NBLK; ++j) {
      int n = col0 + j * 16 + nl;
      float bv = bias[n];
#pragma unroll
      for (int r = 0; r < 8; ++r)
        C[(size_t)(row0 + i * 16 + m0 + r) * NH3 + n] = acc[i][j][r] + bv;
    }
}

// ---------------------------------------------------------------------------
// Persistent BiLSTM recurrence (one layer, both directions).
// blockIdx.x = dir (0 fwd, 1 bwd), 256 threads = 8 waves.
// z = Xg[t] + h @ Wh ; i=sig, c=(1-i)c + i tanh(j), h=tanh(c) sig(o)
// k-outer loop keeps 9 WMMA accumulation chains (3 gate-groups x 3 gates)
// live per wave: one LDS A-fragment feeds 9 WMMAs per k-step. Group map:
// wave 0 -> {0,1,2} + solo pass for {3}; wave w>0 -> {3w+1,3w+2,3w+3}.
// h double-buffered in LDS, one workgroup barrier per timestep.
// ---------------------------------------------------------------------------
__global__ __launch_bounds__(256, 1)
void k_lstm_layer(const float* __restrict__ Xg_f, const float* __restrict__ Xg_b,
                  const _Float16* __restrict__ Wht_f, const _Float16* __restrict__ Wht_b,
                  const float* __restrict__ h0, const float* __restrict__ c0,
                  float* __restrict__ outf, _Float16* __restrict__ out16) {
  __shared__ _Float16 hbuf[2 * NB * NHP];  // double-buffered h (f16, K-padded)
  __shared__ float    cbuf[NB * NH];       // c state
  const int dir = blockIdx.x;
  const float*    Xg  = dir ? Xg_b  : Xg_f;
  const _Float16* Wht = dir ? Wht_b : Wht_f;

  for (int idx = threadIdx.x; idx < NB * NHP; idx += blockDim.x) {
    int k = idx % NHP;
    _Float16 v = (k < NH) ? (_Float16)h0[k] : (_Float16)0.f;
    hbuf[idx] = v;
    hbuf[NB * NHP + idx] = v;
  }
  for (int idx = threadIdx.x; idx < NB * NH; idx += blockDim.x)
    cbuf[idx] = c0[idx % NH];
  __syncthreads();

  const int wave = threadIdx.x >> 5;
  const int lane = threadIdx.x & 31;
  const int koff = (lane >> 4) << 3;
  const int nloc = lane & 15;
  const int m0   = (lane >> 4) << 3;
  int cur = 0;

  // gate epilogue for one 16-column group at base column n0
  auto gate_ep = [&](int t, int n0, v8f& zi, v8f& zj, v8f& zo, int curbuf) {
#pragma unroll
    for (int r = 0; r < 8; ++r) {
      int m = m0 + r;
      int n = n0 + nloc;
      float ig = sigm_(zi[r]);
      float cc = (1.f - ig) * cbuf[m * NH + n] + ig * tanh_(zj[r]);
      float hh = tanh_(cc) * sigm_(zo[r]);
      cbuf[m * NH + n] = cc;
      size_t oidx = ((size_t)m * NT + t) * NH2 + dir * NH + n;
      outf[oidx]  = hh;
      out16[oidx] = (_Float16)hh;
      hbuf[(curbuf ^ 1) * NB * NHP + m * NHP + n] = (_Float16)hh;
    }
  };

  for (int step = 0; step < NT; ++step) {
    const int t = dir ? (NT - 1 - step) : step;
    const _Float16* hA = hbuf + cur * NB * NHP + nloc * NHP + koff;

    // ---- phase A: 3 contiguous groups, 9 accumulation chains -------------
    const int gbase = wave ? (wave * 3 + 1) : 0;
    v8f acc[3][3];   // [group][gate]
#pragma unroll
    for (int q = 0; q < 3; ++q)
#pragma unroll
      for (int gate = 0; gate < 3; ++gate)
#pragma unroll
        for (int r = 0; r < 8; ++r)
          acc[q][gate][r] =
              Xg[((size_t)(m0 + r) * NT + t) * NH3 + gate * NH + (gbase + q) * 16 + nloc];
    const _Float16* bb = Wht + (size_t)(gbase * 16 + nloc) * NHP + koff;
#pragma unroll 1
    for (int k0 = 0; k0 < NHP; k0 += 32) {
      v16h a = ldh16(hA + k0);
#pragma unroll
      for (int q = 0; q < 3; ++q)
#pragma unroll
        for (int gate = 0; gate < 3; ++gate) {
          v16h b = ldh16(bb + (gate * NH + q * 16) * NHP + k0);
          acc[q][gate] = __builtin_amdgcn_wmma_f32_16x16x32_f16(false, a, false, b,
                                                                (short)0, acc[q][gate], false, false);
        }
    }
#pragma unroll
    for (int q = 0; q < 3; ++q)
      gate_ep(t, (gbase + q) * 16, acc[q][0], acc[q][1], acc[q][2], cur);

    // ---- phase B: wave 0 handles the 25th group (g = 3) ------------------
    if (wave == 0) {
      constexpr int g = 3;
      v8f ai, aj, ao;
#pragma unroll
      for (int r = 0; r < 8; ++r) {
        size_t base = ((size_t)(m0 + r) * NT + t) * NH3;
        ai[r] = Xg[base + g * 16 + nloc];
        aj[r] = Xg[base + NH + g * 16 + nloc];
        ao[r] = Xg[base + 2 * NH + g * 16 + nloc];
      }
      const _Float16* bi = Wht + (size_t)(g * 16 + nloc) * NHP + koff;
#pragma unroll 1
      for (int k0 = 0; k0 < NHP; k0 += 32) {
        v16h a = ldh16(hA + k0);
        ai = __builtin_amdgcn_wmma_f32_16x16x32_f16(false, a, false, ldh16(bi + k0), (short)0, ai, false, false);
        aj = __builtin_amdgcn_wmma_f32_16x16x32_f16(false, a, false, ldh16(bi + NH * NHP + k0), (short)0, aj, false, false);
        ao = __builtin_amdgcn_wmma_f32_16x16x32_f16(false, a, false, ldh16(bi + 2 * NH * NHP + k0), (short)0, ao, false, false);
      }
      gate_ep(t, g * 16, ai, aj, ao, cur);
    }

    __syncthreads();
    cur ^= 1;
  }
}

// ---------------------------------------------------------------------------
// Highway: gate = sigmoid(out1 @ W_hw + b); cur16 = f16(g*out1 + (1-g)*out0)
// 4x2 register blocking.
// ---------------------------------------------------------------------------
__global__ __launch_bounds__(256, 1)
void k_gemm_highway(const _Float16* __restrict__ A, const _Float16* __restrict__ Bt,
                    const float* __restrict__ bias, const float* __restrict__ out1f,
                    const float* __restrict__ out0f, _Float16* __restrict__ cur16) {
  constexpr int MB = 4, NBLK = 2;
  constexpr int tn = NH2 / (16 * NBLK);             // 25
  int wave = blockIdx.x * (blockDim.x >> 5) + (threadIdx.x >> 5);
  int row0 = (wave / tn) * (16 * MB);
  int col0 = (wave % tn) * (16 * NBLK);
  v8f acc[MB][NBLK] = {};
  wmma_block<MB, NBLK, NH2, NH2>(A, row0, Bt, col0, NH2, acc);
  int lane = threadIdx.x & 31;
  int nl = lane & 15;
  int m0 = (lane >> 4) << 3;
#pragma unroll
  for (int i = 0; i < MB; ++i)
#pragma unroll
    for (int j = 0; j < NBLK; ++j) {
      int n = col0 + j * 16 + nl;
      float bv = bias[n];
#pragma unroll
      for (int r = 0; r < 8; ++r) {
        size_t idx = (size_t)(row0 + i * 16 + m0 + r) * NH2 + n;
        float g = sigm_(acc[i][j][r] + bv);
        cur16[idx] = (_Float16)(g * out1f[idx] + (1.f - g) * out0f[idx]);
      }
    }
}

// ---------------------------------------------------------------------------
// Projection: s/e = cur @ W + b, bias-augment col 150 = 1, cols>150 = 0.
// Output f16 [BTOT, 160]. 4x1 blocking.
// ---------------------------------------------------------------------------
__global__ __launch_bounds__(256, 1)
void k_gemm_proj(const _Float16* __restrict__ A, const _Float16* __restrict__ Bt,
                 const float* __restrict__ bias, _Float16* __restrict__ C) {
  constexpr int MB = 4, NBLK = 1;
  constexpr int tn = NFP / 16;                      // 10
  int wave = blockIdx.x * (blockDim.x >> 5) + (threadIdx.x >> 5);
  int row0 = (wave / tn) * (16 * MB);
  int col0 = (wave % tn) * 16;
  v8f acc[MB][NBLK] = {};
  wmma_block<MB, NBLK, NH2, NH2>(A, row0, Bt, col0, NH2, acc);
  int lane = threadIdx.x & 31;
  int n = col0 + (lane & 15);
  int m0 = (lane >> 4) << 3;
  float bv = (n < NF) ? bias[n] : 0.f;
#pragma unroll
  for (int i = 0; i < MB; ++i)
#pragma unroll
    for (int r = 0; r < 8; ++r) {
      float v = (n < NF) ? (acc[i][0][r] + bv) : ((n == NF) ? 1.f : 0.f);
      C[(size_t)(row0 + i * 16 + m0 + r) * NFP + n] = (_Float16)v;
    }
}

// ---------------------------------------------------------------------------
// Biaffine stage 1: left[c] = s1 @ U[:,c,:]   (f16 out [NC][BTOT,160]),
// 4x2 blocking.
// ---------------------------------------------------------------------------
__global__ __launch_bounds__(256, 1)
void k_gemm_left(const _Float16* __restrict__ s1, const _Float16* __restrict__ Ut,
                 _Float16* __restrict__ left) {
  constexpr int MB = 4, NBLK = 2;
  constexpr int tiles_per_c = (BTOT / (16 * MB)) * (NFP / (16 * NBLK));  // 64*5
  int wave = blockIdx.x * (blockDim.x >> 5) + (threadIdx.x >> 5);
  int c = wave / tiles_per_c;
  int rem = wave % tiles_per_c;
  int row0 = (rem / (NFP / (16 * NBLK))) * (16 * MB);
  int col0 = (rem % (NFP / (16 * NBLK))) * (16 * NBLK);
  v8f acc[MB][NBLK] = {};
  wmma_block<MB, NBLK, NFP, NFP>(s1, row0, Ut + (size_t)c * NFP * NFP, col0, NFP, acc);
  int lane = threadIdx.x & 31;
  int nl = lane & 15;
  int m0 = (lane >> 4) << 3;
  _Float16* out = left + (size_t)c * BTOT * NFP;
#pragma unroll
  for (int i = 0; i < MB; ++i)
#pragma unroll
    for (int j = 0; j < NBLK; ++j) {
      int n = col0 + j * 16 + nl;
#pragma unroll
      for (int r = 0; r < 8; ++r)
        out[(size_t)(row0 + i * 16 + m0 + r) * NFP + n] = (_Float16)acc[i][j][r];
    }
}

// ---------------------------------------------------------------------------
// Biaffine stage 2: out[b,x,y,c] = left[c][b*256+x,:] . e1[b*256+y,:]
// e1 rows ARE the transposed-B operand. 2x2 blocking.
// ---------------------------------------------------------------------------
__global__ __launch_bounds__(256, 1)
void k_biaffine(const _Float16* __restrict__ left, const _Float16* __restrict__ e1,
                float* __restrict__ outp) {
  constexpr int MB = 2, NBLK = 2;
  int wave = blockIdx.x * (blockDim.x >> 5) + (threadIdx.x >> 5);
  int yt = wave & 7;          // 8 y-blocks of 32
  int xt = (wave >> 3) & 7;   // 8 x-blocks of 32
  int c  = (wave >> 6) & 7;
  int b  = wave >> 9;
  const _Float16* A  = left + ((size_t)c * BTOT + (size_t)b * NT) * NFP;
  const _Float16* Bt = e1 + (size_t)b * NT * NFP;
  v8f acc[MB][NBLK] = {};
  wmma_block<MB, NBLK, NFP, NFP>(A, xt * 32, Bt, yt * 32, NFP, acc);
  int lane = threadIdx.x & 31;
  int nl = lane & 15;
  int m0 = (lane >> 4) << 3;
#pragma unroll
  for (int i = 0; i < MB; ++i)
#pragma unroll
    for (int j = 0; j < NBLK; ++j) {
      int y  = yt * 32 + j * 16 + nl;
      int x0 = xt * 32 + i * 16 + m0;
#pragma unroll
      for (int r = 0; r < 8; ++r)
        outp[(((size_t)b * NT + (x0 + r)) * NT + y) * NC + c] = acc[i][j][r];
    }
}

// ---------------------------------------------------------------------------
// Host orchestration
// ---------------------------------------------------------------------------
extern "C" void kernel_launch(void* const* d_in, const int* in_sizes, int n_in,
                              void* d_out, int out_size, void* d_ws, size_t ws_size,
                              hipStream_t stream) {
  (void)in_sizes; (void)n_in; (void)out_size; (void)ws_size;
  const float* x    = (const float*)d_in[0];
  const float* W_f0 = (const float*)d_in[1];  const float* b_f0 = (const float*)d_in[2];
  const float* W_b0 = (const float*)d_in[3];  const float* b_b0 = (const float*)d_in[4];
  const float* W_f1 = (const float*)d_in[5];  const float* b_f1 = (const float*)d_in[6];
  const float* W_b1 = (const float*)d_in[7];  const float* b_b1 = (const float*)d_in[8];
  const float* h0   = (const float*)d_in[9];  const float* c0   = (const float*)d_in[10];
  const float* W_hw = (const float*)d_in[11]; const float* b_hw = (const float*)d_in[12];
  const float* W_s  = (const float*)d_in[13]; const float* b_s  = (const float*)d_in[14];
  const float* W_e  = (const float*)d_in[15]; const float* b_e  = (const float*)d_in[16];
  const float* U    = (const float*)d_in[17];
  float* outp = (float*)d_out;

  char* ws = (char*)d_ws;
  size_t off = 0;
  auto take = [&](size_t bytes) -> void* {
    void* p = ws + off; off += (bytes + 255) & ~(size_t)255; return p;
  };
  _Float16* x16     = (_Float16*)take((size_t)BTOT * ND * 2);
  _Float16* Wxt_f0  = (_Float16*)take((size_t)NH3 * ND * 2);
  _Float16* Wxt_b0  = (_Float16*)take((size_t)NH3 * ND * 2);
  _Float16* Wht_f0  = (_Float16*)take((size_t)NH3 * NHP * 2);
  _Float16* Wht_b0  = (_Float16*)take((size_t)NH3 * NHP * 2);
  _Float16* Wxt_f1  = (_Float16*)take((size_t)NH3 * NH2 * 2);
  _Float16* Wxt_b1  = (_Float16*)take((size_t)NH3 * NH2 * 2);
  _Float16* Wht_f1  = (_Float16*)take((size_t)NH3 * NHP * 2);
  _Float16* Wht_b1  = (_Float16*)take((size_t)NH3 * NHP * 2);
  _Float16* Whwt    = (_Float16*)take((size_t)NH2 * NH2 * 2);
  _Float16* Wst     = (_Float16*)take((size_t)NFP * NH2 * 2);
  _Float16* Wet     = (_Float16*)take((size_t)NFP * NH2 * 2);
  _Float16* Ut      = (_Float16*)take((size_t)NC * NFP * NFP * 2);
  float*    Xg_f    = (float*)take((size_t)BTOT * NH3 * 4);   // reused layer 0/1
  float*    Xg_b    = (float*)take((size_t)BTOT * NH3 * 4);
  float*    out0    = (float*)take((size_t)BTOT * NH2 * 4);
  _Float16* out0_16 = (_Float16*)take((size_t)BTOT * NH2 * 2);
  float*    out1    = (float*)take((size_t)BTOT * NH2 * 4);
  _Float16* out1_16 = (_Float16*)take((size_t)BTOT * NH2 * 2);
  _Float16* cur16   = (_Float16*)take((size_t)BTOT * NH2 * 2);
  _Float16* s1      = (_Float16*)take((size_t)BTOT * NFP * 2);
  _Float16* e1      = (_Float16*)take((size_t)BTOT * NFP * 2);
  _Float16* left    = (_Float16*)take((size_t)NC * BTOT * NFP * 2);

  const int TPB = 256;
  auto cdiv = [](int a, int b) { return (a + b - 1) / b; };

  // --- precision conversions / weight transposes -------------------------
  k_f32_to_f16<<<cdiv(BTOT * ND, TPB), TPB, 0, stream>>>(x, x16, BTOT * ND);
  k_transpose_w<<<cdiv(NH3 * ND, TPB), TPB, 0, stream>>>(W_f0, Wxt_f0, 0, ND, ND, NH3, NH3, NH3);
  k_transpose_w<<<cdiv(NH3 * ND, TPB), TPB, 0, stream>>>(W_b0, Wxt_b0, 0, ND, ND, NH3, NH3, NH3);
  k_transpose_w<<<cdiv(NH3 * NHP, TPB), TPB, 0, stream>>>(W_f0, Wht_f0, ND, NH, NHP, NH3, NH3, NH3);
  k_transpose_w<<<cdiv(NH3 * NHP, TPB), TPB, 0, stream>>>(W_b0, Wht_b0, ND, NH, NHP, NH3, NH3, NH3);
  k_transpose_w<<<cdiv(NH3 * NH2, TPB), TPB, 0, stream>>>(W_f1, Wxt_f1, 0, NH2, NH2, NH3, NH3, NH3);
  k_transpose_w<<<cdiv(NH3 * NH2, TPB), TPB, 0, stream>>>(W_b1, Wxt_b1, 0, NH2, NH2, NH3, NH3, NH3);
  k_transpose_w<<<cdiv(NH3 * NHP, TPB), TPB, 0, stream>>>(W_f1, Wht_f1, NH2, NH, NHP, NH3, NH3, NH3);
  k_transpose_w<<<cdiv(NH3 * NHP, TPB), TPB, 0, stream>>>(W_b1, Wht_b1, NH2, NH, NHP, NH3, NH3, NH3);
  k_transpose_w<<<cdiv(NH2 * NH2, TPB), TPB, 0, stream>>>(W_hw, Whwt, 0, NH2, NH2, NH2, NH2, NH2);
  k_transpose_w<<<cdiv(NFP * NH2, TPB), TPB, 0, stream>>>(W_s, Wst, 0, NH2, NH2, NF, NFP, NF);
  k_transpose_w<<<cdiv(NFP * NH2, TPB), TPB, 0, stream>>>(W_e, Wet, 0, NH2, NH2, NF, NFP, NF);
  k_transpose_U<<<cdiv(NC * NFP * NFP, TPB), TPB, 0, stream>>>(U, Ut);

  // --- layer 0: batched input GEMMs + persistent recurrence --------------
  // waves = (4096/64)*(1200/48) = 1600 -> 200 blocks of 8 waves
  const int xg_blocks = (BTOT / 64) * (NH3 / 48) / 8;
  k_gemm_bias_f32<ND, ND><<<xg_blocks, TPB, 0, stream>>>(x16, Wxt_f0, b_f0, Xg_f);
  k_gemm_bias_f32<ND, ND><<<xg_blocks, TPB, 0, stream>>>(x16, Wxt_b0, b_b0, Xg_b);
  k_lstm_layer<<<2, TPB, 0, stream>>>(Xg_f, Xg_b, Wht_f0, Wht_b0, h0, c0, out0, out0_16);

  // --- layer 1 (reuse Xg buffers) ----------------------------------------
  k_gemm_bias_f32<NH2, NH2><<<xg_blocks, TPB, 0, stream>>>(out0_16, Wxt_f1, b_f1, Xg_f);
  k_gemm_bias_f32<NH2, NH2><<<xg_blocks, TPB, 0, stream>>>(out0_16, Wxt_b1, b_b1, Xg_b);
  k_lstm_layer<<<2, TPB, 0, stream>>>(Xg_f, Xg_b, Wht_f1, Wht_b1, h0, c0, out1, out1_16);

  // --- highway gating (fused GEMM + blend -> cur16) ----------------------
  // waves = (4096/64)*(800/32) = 1600 -> 200 blocks
  k_gemm_highway<<<(BTOT / 64) * (NH2 / 32) / 8, TPB, 0, stream>>>(
      out1_16, Whwt, b_hw, out1, out0, cur16);

  // --- start/end projections with bias-augment column --------------------
  // waves = (4096/64)*(160/16) = 640 -> 80 blocks
  const int proj_blocks = (BTOT / 64) * (NFP / 16) / 8;
  k_gemm_proj<<<proj_blocks, TPB, 0, stream>>>(cur16, Wst, b_s, s1);
  k_gemm_proj<<<proj_blocks, TPB, 0, stream>>>(cur16, Wet, b_e, e1);

  // --- biaffine ----------------------------------------------------------
  // stage1: waves = 8*(4096/64)*(160/32) = 2560 -> 320 blocks
  k_gemm_left<<<NC * (BTOT / 64) * (NFP / 32) / 8, TPB, 0, stream>>>(s1, Ut, left);
  // stage2: waves = 16*8*8*8 = 8192 -> 1024 blocks
  k_biaffine<<<NB * NC * (NT / 32) * (NT / 32) / 8, TPB, 0, stream>>>(left, e1, outp);
}
